// MHNNSConv_40458591928749
// MI455X (gfx1250) — compile-verified, matrix operand order checked by reading, not run on hardware
//
#include <hip/hip_runtime.h>
#include <hip/hip_bf16.h>

// ---------------- problem constants (match reference) ----------------
#define NV   50000   // nodes
#define NE   20000   // hyperedges
#define NNZ  400000  // incidence entries
#define DD   256     // hid_dim
#define ALPHA 0.5f

typedef __attribute__((ext_vector_type(16))) __bf16 v16bf;
typedef __attribute__((ext_vector_type(8)))  float  v8f;

// Hardware fp32 global atomic add (non-returning): GLOBAL_ATOMIC_ADD_F32.
// Guarantees the single-instruction CDNA5 atomic path (no CAS loop).
__device__ __forceinline__ void atomic_add_f32_hw(float* p, float v) {
    asm volatile("global_atomic_add_f32 %0, %1, off" :: "v"(p), "v"(v) : "memory");
}

// ============================================================================
// Weight pre-pack: fp32 row-major [K][256] -> bf16 fragment-ready layout.
// Fragment (kkTile, nTile): lane L holds 16 contiguous bf16 at
//   P[(((kkTile*16) + nTile)*32 + L)*16 + j]
// matching the wave32 WMMA B layout (n = nTile*16 + (L&15), g = L>>4,
// j=2v{+1}: k = kkTile*32 + (v<4 ? g*8+2v : 16+g*8+2(v-4)) {+1}).
// ============================================================================
__global__ void __launch_bounds__(256)
pack_weights_bf16(const float* __restrict__ W, __bf16* __restrict__ P, int K) {
    const int idx = blockIdx.x * 256 + threadIdx.x;   // one bf16 element each
    if (idx >= K * DD) return;
    const int j      = idx & 15;
    const int lane   = (idx >> 4) & 31;
    const int nTile  = (idx >> 9) & 15;
    const int kkTile = idx >> 13;
    const int g  = lane >> 4;
    const int n  = nTile * 16 + (lane & 15);
    const int v  = j >> 1;
    const int lo = j & 1;
    const int k  = kkTile * 32 +
                   ((v < 4) ? (g * 8 + 2 * v) : (16 + g * 8 + 2 * (v - 4))) + lo;
    P[idx] = (__bf16)W[(size_t)k * DD + n];
}

// ---------------- A-fragment: 16x32 bf16 from fp32 row-major ----------------
// Contiguous per half: elems 0..7 = A[row][kk + g*8 .. +7],
//                      elems 8..15 = A[row][kk + 16 + g*8 .. +7]
__device__ __forceinline__ v16bf load_afrag_f32(const float* __restrict__ A,
                                                int row, int kk, int lane) {
    const int g = lane >> 4;
    const float* ab = A + (size_t)row * DD + kk + g * 8;
    float4 q0 = *reinterpret_cast<const float4*>(ab);
    float4 q1 = *reinterpret_cast<const float4*>(ab + 4);
    float4 q2 = *reinterpret_cast<const float4*>(ab + 16);
    float4 q3 = *reinterpret_cast<const float4*>(ab + 20);
    v16bf a;
    a[0]=(__bf16)q0.x;  a[1]=(__bf16)q0.y;  a[2]=(__bf16)q0.z;  a[3]=(__bf16)q0.w;
    a[4]=(__bf16)q1.x;  a[5]=(__bf16)q1.y;  a[6]=(__bf16)q1.z;  a[7]=(__bf16)q1.w;
    a[8]=(__bf16)q2.x;  a[9]=(__bf16)q2.y;  a[10]=(__bf16)q2.z; a[11]=(__bf16)q2.w;
    a[12]=(__bf16)q3.x; a[13]=(__bf16)q3.y; a[14]=(__bf16)q3.z; a[15]=(__bf16)q3.w;
    return a;
}

// K=256 panel: acc[0..7] += A(16 rows) @ B(256 x 128-col slab).
// All 8 B-fragments of a K-step are preloaded into registers first so the
// scheduler can issue one load clause + partial loadcnt waits and keep the
// WMMA pipe busy (instead of a full s_wait_loadcnt 0 per WMMA).
__device__ __forceinline__ void mma_panel(const float* __restrict__ A,
                                          const v16bf* __restrict__ B,
                                          int aRow, int lane, int nT0,
                                          v8f acc[8]) {
#pragma unroll
    for (int kk = 0; kk < DD; kk += 32) {
        v16bf a = load_afrag_f32(A, aRow, kk, lane);
        const v16bf* bb = B + ((size_t)(kk >> 5) * 16 + nT0) * 32 + lane;
        v16bf b[8];
#pragma unroll
        for (int t = 0; t < 8; ++t) b[t] = bb[t * 32];
#pragma unroll
        for (int t = 0; t < 8; ++t) {
            acc[t] = __builtin_amdgcn_wmma_f32_16x16x32_bf16(
                false, a, false, b[t], (short)0, acc[t], false, false);
        }
    }
}

// ============================================================================
// Fused GEMM: C = A1@B1 [+ A2@B2] + bias, optional mask + residual mix.
// grid: x = ceil(rowTiles/8), y = 2 (128-col slabs); block = 256 (8 waves).
// Each wave computes a 16x128 output tile (8 WMMA accumulators).
// mode 0: C = acc + bias[n]
// mode 1: v = acc + bias[n]; v = (mask[m]>0)? v : 0;
//         C = (1-ALPHA)*v + ALPHA*X0[m][n]
// ============================================================================
__global__ void __launch_bounds__(256)
gemm_bf16_wmma(const float* __restrict__ A1, const v16bf* __restrict__ B1,
               const float* __restrict__ A2, const v16bf* __restrict__ B2,
               const float* __restrict__ bias,
               const float* __restrict__ mask,
               const float* __restrict__ X0,
               float* __restrict__ C,
               int nRowTiles, int mode) {
    const int lane = threadIdx.x & 31;
    const int wave = threadIdx.x >> 5;
    const int rowTile = blockIdx.x * 8 + wave;
    if (rowTile >= nRowTiles) return;           // wave-uniform: EXEC stays all-1s

    const int nT0     = blockIdx.y * 8;         // first 16-col tile of this slab
    const int colBase = nT0 * 16;
    const int aRow    = rowTile * 16 + (lane & 15);
    const int n0      = lane & 15;

    v8f acc[8];
#pragma unroll
    for (int t = 0; t < 8; ++t) acc[t] = (v8f)(0.0f);

    mma_panel(A1, B1, aRow, lane, nT0, acc);
    if (A2 != nullptr) mma_panel(A2, B2, aRow, lane, nT0, acc);

    // C/D layout: VGPR r -> m = r + (lane>>4)*8 ; n = lane&15
    const int mBase = rowTile * 16 + ((lane >> 4) * 8);
#pragma unroll
    for (int r = 0; r < 8; ++r) {
        const int m = mBase + r;
#pragma unroll
        for (int t = 0; t < 8; ++t) {
            const int c = colBase + t * 16 + n0;
            float v = acc[t][r] + bias[c];
            if (mode == 1) {
                v = (mask[m] > 0.0f) ? v : 0.0f;
                v = (1.0f - ALPHA) * v + ALPHA * X0[(size_t)m * DD + c];
            }
            C[(size_t)m * DD + c] = v;
        }
    }
}

// ============================================================================
// Sparse phase. 4 incidence entries per 256-thread block; 64 lanes per entry,
// each lane moves 4 consecutive floats (b128 load + 4 hw f32 atomics).
// ============================================================================
__global__ void __launch_bounds__(256)
scatter_rows(const float* __restrict__ src,   // [rowsSrc][256] gather source
             const int* __restrict__ gidx,    // gather index per entry
             const int* __restrict__ sidx,    // scatter index per entry
             float* __restrict__ dstSum,      // [rowsDst][256] accumulators
             float* __restrict__ cnt) {       // [rowsDst] counts
    const int slot = threadIdx.x >> 6;        // 0..3: entry within block
    const int l    = threadIdx.x & 63;        // lane within entry
    const int i    = blockIdx.x * 4 + slot;
    if (i >= NNZ) return;
    const int gs = gidx[i];
    const int ss = sidx[i];
    float4 f = *reinterpret_cast<const float4*>(src + (size_t)gs * DD + l * 4);
    float* d = dstSum + (size_t)ss * DD + l * 4;
    atomic_add_f32_hw(d + 0, f.x);
    atomic_add_f32_hw(d + 1, f.y);
    atomic_add_f32_hw(d + 2, f.z);
    atomic_add_f32_hw(d + 3, f.w);
    if (l == 0) atomic_add_f32_hw(&cnt[ss], 1.0f);
}

// dat[row][:] /= max(cnt[row], 1)   -- 4 rows per block, float4 per lane
__global__ void __launch_bounds__(256)
row_normalize(float* __restrict__ dat, const float* __restrict__ cnt) {
    const int row = blockIdx.x * 4 + (threadIdx.x >> 6);
    const int l   = threadIdx.x & 63;
    const float inv = 1.0f / fmaxf(cnt[row], 1.0f);
    float4* p = reinterpret_cast<float4*>(dat + (size_t)row * DD + l * 4);
    float4 f = *p;
    f.x *= inv; f.y *= inv; f.z *= inv; f.w *= inv;
    *p = f;
}

// ---------------- launch ----------------
extern "C" void kernel_launch(void* const* d_in, const int* in_sizes, int n_in,
                              void* d_out, int out_size, void* d_ws, size_t ws_size,
                              hipStream_t stream) {
    const float* X  = (const float*)d_in[0];
    const float* X0 = (const float*)d_in[1];
    const int*   vertex = (const int*)d_in[2];
    const int*   edges  = (const int*)d_in[3];
    const float* W1 = (const float*)d_in[4];
    const float* b1 = (const float*)d_in[5];
    const float* W2 = (const float*)d_in[6];
    const float* b2 = (const float*)d_in[7];
    const float* W3 = (const float*)d_in[8];
    const float* b3 = (const float*)d_in[9];
    float* out = (float*)d_out;

    char* ws = (char*)d_ws;
    const size_t offXW1 = 0;                                  // N*D f32 (reused as Xout)
    const size_t offXe  = offXW1 + (size_t)NV * DD * 4;
    const size_t offAg  = offXe  + (size_t)NE * DD * 4;
    const size_t offEc  = offAg  + (size_t)NV * DD * 4;
    const size_t offVc  = offEc  + (size_t)NE * 4;
    const size_t offW1p = offVc  + (size_t)NV * 4;
    const size_t offW2p = offW1p + (size_t)DD * DD * 2;
    const size_t offW3p = offW2p + (size_t)2 * DD * DD * 2;

    float*  XW1  = (float*)(ws + offXW1);     // also Xout after the mode-1 GEMM
    float*  Xe   = (float*)(ws + offXe);
    float*  Ag   = (float*)(ws + offAg);
    float*  ecnt = (float*)(ws + offEc);
    float*  vcnt = (float*)(ws + offVc);
    __bf16* W1p  = (__bf16*)(ws + offW1p);
    __bf16* W2p  = (__bf16*)(ws + offW2p);
    __bf16* W3p  = (__bf16*)(ws + offW3p);

    // zero accumulators + counts (contiguous region)
    hipMemsetAsync(ws + offXe, 0, offW1p - offXe, stream);

    // pack weights into bf16 fragment-ready layout
    pack_weights_bf16<<<(DD * DD + 255) / 256, 256, 0, stream>>>(W1, W1p, DD);
    pack_weights_bf16<<<(2 * DD * DD + 255) / 256, 256, 0, stream>>>(W2, W2p, 2 * DD);
    pack_weights_bf16<<<(DD * DD + 255) / 256, 256, 0, stream>>>(W3, W3p, DD);

    const int rowTiles = NV / 16;                 // 3125
    dim3 ggrid((rowTiles + 7) / 8, 2);            // (391, 2): two 128-col slabs

    const v16bf* W1v = (const v16bf*)W1p;
    const v16bf* W2v = (const v16bf*)W2p;         // K rows 0..255 of W2
    const v16bf* W2v_hi = W2v + (size_t)(DD * DD) / 16;   // K rows 256..511
    const v16bf* W3v = (const v16bf*)W3p;

    // 1) XW1 = X @ W1 + b1
    gemm_bf16_wmma<<<ggrid, 256, 0, stream>>>(X, W1v, nullptr, nullptr, b1,
                                              nullptr, nullptr, XW1, rowTiles, 0);
    // 2) Xe = scatter_mean(XW1[vertex], edges)   (gather by vertex, scatter by edge)
    scatter_rows<<<NNZ / 4, 256, 0, stream>>>(XW1, vertex, edges, Xe, ecnt);
    row_normalize<<<NE / 4, 256, 0, stream>>>(Xe, ecnt);
    // 3) Ag = scatter_mean(Xe[edges], vertex)    (gather by edge, scatter by vertex)
    scatter_rows<<<NNZ / 4, 256, 0, stream>>>(Xe, edges, vertex, Ag, vcnt);
    row_normalize<<<NV / 4, 256, 0, stream>>>(Ag, vcnt);
    // 4) Xout = (1-a)*mask*(X@W2a + Ag@W2b + b2) + a*X0   (reuse XW1 buffer)
    gemm_bf16_wmma<<<ggrid, 256, 0, stream>>>(X, W2v, Ag, W2v_hi, b2,
                                              vcnt, X0, XW1, rowTiles, 1);
    // 5) out = Xout @ W3 + b3
    gemm_bf16_wmma<<<ggrid, 256, 0, stream>>>(XW1, W3v, nullptr, nullptr, b3,
                                              nullptr, nullptr, out, rowTiles, 0);
    (void)in_sizes; (void)n_in; (void)out_size; (void)ws_size;
}
